// PerformerClassifier_16509854285933
// MI455X (gfx1250) — matrix operand order repaired and checked by reading, not compiled, not run on hardware
//
#include <hip/hip_runtime.h>
#include <math.h>

typedef __attribute__((ext_vector_type(16))) __bf16 v16bf;
typedef __attribute__((ext_vector_type(8)))  __bf16 v8bf;
typedef __attribute__((ext_vector_type(8)))  float  v8f;

namespace {
constexpr int   Bn = 8, Ng = 16384, Dm = 64, Hh = 2, Dh = 32, Fq = 110, Fp = 128, Dff = 256;
constexpr int   TOKENS = Bn * Ng;      // 131072
constexpr int   TOKT   = TOKENS / 16;  // 8192 token tiles
constexpr float DN     = 0.42044820762685725f;  // 32^-0.25
constexpr float DN2    = 0.17677669529663687f;  // 32^-0.5
constexpr float RATIO  = 0.09534625892455924f;  // 110^-0.5
constexpr float FEPS   = 1e-4f;
}

// ---------------- fragment helpers (CDNA5 wave32 WMMA layouts) ----------------
__device__ __forceinline__ int c_row(int r){ return ((threadIdx.x & 31) >> 4) ? r + 8 : r; }
__device__ __forceinline__ int c_col()    { return threadIdx.x & 15; }

__device__ __forceinline__ v8f zero8(){
  v8f c;
#pragma unroll
  for (int i = 0; i < 8; ++i) c[i] = 0.f;
  return c;
}

__device__ __forceinline__ v16bf cat8(v8bf lo, v8bf hi){
  v16bf a;
#pragma unroll
  for (int j = 0; j < 8; ++j){ a[j] = lo[j]; a[8+j] = hi[j]; }
  return a;
}

__device__ __forceinline__ v8f wmma_bf16(v16bf a, v16bf b, v8f c){
  return __builtin_amdgcn_wmma_f32_16x16x32_bf16(false, a, false, b, (short)0, c, false, false);
}

// A fragment: 16(M) x 32(K), row-major source. Each lane holds one contiguous
// 16-element K-run (k = half*16 + j) -> two 16B vector loads.
__device__ __forceinline__ v16bf load_a_bf16(const __bf16* A, int ld){
  const int lane = threadIdx.x & 31, half = lane >> 4, m = lane & 15;
  const __bf16* p = A + m*ld + half*16;
  return cat8(*(const v8bf*)p, *(const v8bf*)(p + 8));
}

// B fragment from pre-swizzled tile (lane-major, 16 contiguous bf16 per lane)
__device__ __forceinline__ v16bf load_b_sw(const __bf16* T){
  const __bf16* p = T + (threadIdx.x & 31)*16;
  return cat8(*(const v8bf*)p, *(const v8bf*)(p + 8));
}

// B fragment gather: 32(K) x 16(N), row-major source (used only for v in k_ctx)
__device__ __forceinline__ v16bf load_b_bf16(const __bf16* Bm, int ld){
  const int lane = threadIdx.x & 31, half = lane >> 4, n = lane & 15;
  v16bf b;
#pragma unroll
  for (int j = 0; j < 8; ++j){
    int k0 = half*16 + 2*j;
    b[2*j]   = Bm[k0*ld + n];
    b[2*j+1] = Bm[(k0+1)*ld + n];
  }
  return b;
}

// order-preserving float<->uint encoding for atomicMax
__device__ __forceinline__ unsigned fenc(float f){
  unsigned u = __float_as_uint(f);
  return (u & 0x80000000u) ? ~u : (u | 0x80000000u);
}
__device__ __forceinline__ float fdec(unsigned u){
  unsigned v = (u & 0x80000000u) ? (u & 0x7fffffffu) : ~u;
  return __uint_as_float(v);
}

__device__ __forceinline__ unsigned pack_bf16(float lo, float hi){
  unsigned short ul = __builtin_bit_cast(unsigned short, (__bf16)lo);
  unsigned short uh = __builtin_bit_cast(unsigned short, (__bf16)hi);
  return (unsigned)ul | ((unsigned)uh << 16);
}

// ---------------- kernels ----------------
__global__ void k_embed(const float* __restrict__ x, const float* __restrict__ embw, float* h){
  int idx = blockIdx.x * blockDim.x + threadIdx.x;   // TOKENS*64 elements
  int j = idx & 63;
  int t = idx >> 6;
  int b = t >> 14, n = t & (Ng - 1);
  h[idx] = (j < Dm - 1) ? embw[n*(Dm-1) + j] : x[(size_t)b*Ng + n];
}

// convert a [L][K][N] fp32 weight into B-fragment-order bf16 tiles
// tile = kt*(N/16)+nt ; element (lane,j): n = nt*16+lane%16, k = kt*32+(lane/16)*16+j
__global__ void k_prep_w(const float* __restrict__ src, __bf16* __restrict__ dst,
                         int K, int N){
  int idx = blockIdx.x * blockDim.x + threadIdx.x;
  int per = K * N;
  int l = idx / per, rem = idx % per;
  int NT = N >> 4;
  int tile = rem >> 9, lane = (rem >> 4) & 31, j = rem & 15;
  int kt = tile / NT, nt = tile % NT;
  int half = lane >> 4, n = nt*16 + (lane & 15);
  int k = kt*32 + half*16 + j;
  dst[idx] = (__bf16)src[(size_t)l*per + k*N + n];
}

// proj[l] (110x32 fp32) -> swizzled bf16 B tiles [ft][lane][j], dn folded, pad zeroed
__global__ void k_prep_proj(const float* __restrict__ proj, __bf16* __restrict__ projS){
  int idx = blockIdx.x * blockDim.x + threadIdx.x;   // 8*512
  int tile = idx >> 9, lane = (idx >> 4) & 31, j = idx & 15;
  int half = lane >> 4, f = tile*16 + (lane & 15);
  int k = half*16 + j;                                // k = d
  projS[idx] = (f < Fq) ? (__bf16)(proj[f*Dh + k] * DN) : (__bf16)0.f;
}

// ctx fp32 [bh][128f][32d] -> swizzled bf16 B tiles per bh: tile = fc*2+dt
__global__ void k_prep_ctx(const float* __restrict__ ctx, __bf16* __restrict__ ctxS){
  int idx = blockIdx.x * blockDim.x + threadIdx.x;   // 16*4096
  int bh = idx >> 12, rem = idx & 4095;
  int tile = rem >> 9, lane = (rem >> 4) & 31, j = rem & 15;
  int fc = tile >> 1, dt = tile & 1;
  int half = lane >> 4, n = dt*16 + (lane & 15);
  int k = fc*32 + half*16 + j;                        // k = f
  ctxS[idx] = (__bf16)ctx[(size_t)bh*Fp*Dh + k*Dh + n];
}

__global__ void k_ln(const float* __restrict__ h, const float* __restrict__ g,
                     const float* __restrict__ bta, __bf16* __restrict__ y){
  int t = blockIdx.x * blockDim.x + threadIdx.x;     // one token per thread
  const float4* hp4 = (const float4*)(h + (size_t)t * Dm);
  float xs[64];
#pragma unroll
  for (int j = 0; j < 16; ++j){
    float4 v = hp4[j];
    xs[4*j] = v.x; xs[4*j+1] = v.y; xs[4*j+2] = v.z; xs[4*j+3] = v.w;
  }
  float mu = 0.f;
#pragma unroll
  for (int j = 0; j < Dm; ++j) mu += xs[j];
  mu *= (1.f / Dm);
  float var = 0.f;
#pragma unroll
  for (int j = 0; j < Dm; ++j){ float d = xs[j] - mu; var += d*d; }
  var *= (1.f / Dm);
  float rs = rsqrtf(var + 1e-5f);
  unsigned* yo = (unsigned*)(y + (size_t)t * Dm);
#pragma unroll
  for (int j = 0; j < 32; ++j){
    float a = (xs[2*j]   - mu) * rs * g[2*j]   + bta[2*j];
    float b = (xs[2*j+1] - mu) * rs * g[2*j+1] + bta[2*j+1];
    yo[j] = pack_bf16(a, b);
  }
}

// fused Q/K/V projection: one wave = 16 tokens x 64 cols of one of {q,k,v}
__global__ void k_qkv(const __bf16* __restrict__ y,
                      const __bf16* wqS, const float* bq,
                      const __bf16* wkS, const float* bk,
                      const __bf16* wvS, const float* bv,
                      __bf16* q, __bf16* k, __bf16* v){
  __shared__ __align__(16) __bf16 st[8][16*64];
  int wslot = threadIdx.x >> 5, lane = threadIdx.x & 31;
  int wid = (blockIdx.x * blockDim.x + threadIdx.x) >> 5;
  int mat = wid / TOKT;
  int t0 = (wid % TOKT) * 16;
  const __bf16* W  = (mat == 0) ? wqS : (mat == 1) ? wkS : wvS;
  const float*  Bi = (mat == 0) ? bq  : (mat == 1) ? bk  : bv;
  __bf16* O = (mat == 0) ? q : (mat == 1) ? k : v;
  v8f c[4] = {zero8(), zero8(), zero8(), zero8()};
#pragma unroll
  for (int kt = 0; kt < 2; ++kt){
    v16bf a = load_a_bf16(y + (size_t)t0*Dm + kt*32, Dm);
#pragma unroll
    for (int nt = 0; nt < 4; ++nt)
      c[nt] = wmma_bf16(a, load_b_sw(W + (kt*4 + nt)*512), c[nt]);
  }
#pragma unroll
  for (int nt = 0; nt < 4; ++nt){
    int col = nt*16 + c_col();
    float bias = Bi[col];
#pragma unroll
    for (int r = 0; r < 8; ++r)
      st[wslot][c_row(r)*64 + col] = (__bf16)(c[nt][r] + bias);
  }
  __syncthreads();
#pragma unroll
  for (int i = 0; i < 4; ++i){
    int seg = i*32 + lane;                  // 128 segments of 8 bf16
    int row = seg >> 3, sc = (seg & 7) * 8;
    int t = t0 + row, b_ = t >> 14, n = t & (Ng - 1);
    int head = sc >> 5, dc = sc & 31;
    uint4 d = *(const uint4*)&st[wslot][row*64 + sc];
    *(uint4*)(O + (((size_t)(b_*Hh + head))*Ng + n)*Dh + dc) = d;
  }
}

__global__ void k_zero_state(float* ctx, float* kpsum, unsigned* mg){
  int i = blockIdx.x * blockDim.x + threadIdx.x;     // 16*Fp*Dh threads
  ctx[i] = 0.f;
  if (i < 16*Fp) kpsum[i] = 0.f;
  if (i == 0) *mg = fenc(-INFINITY);
}

// global max over key features dd = k @ (dn*proj)^T
__global__ void k_dk_max(const __bf16* __restrict__ kb, const __bf16* __restrict__ projS,
                         unsigned* mg){
  int wid = (blockIdx.x * blockDim.x + threadIdx.x) >> 5;
  int bh = wid / (Ng/16), tile = wid % (Ng/16);
  int n0 = tile * 16;
  v16bf a = load_a_bf16(kb + ((size_t)bh*Ng + n0)*Dh, Dh);
  float mx = -INFINITY;
#pragma unroll
  for (int ft = 0; ft < 7; ++ft){
    v8f c = wmma_bf16(a, load_b_sw(projS + ft*512), zero8());
    if (ft*16 + c_col() < Fq){
#pragma unroll
      for (int r = 0; r < 8; ++r) mx = fmaxf(mx, c[r]);
    }
  }
#pragma unroll
  for (int off = 16; off; off >>= 1) mx = fmaxf(mx, __shfl_xor(mx, off, 32));
  if ((threadIdx.x & 31) == 0) atomicMax(mg, fenc(mx));
}

// fused kp features + kp_sum + ctx += kp^T @ v   (32 tokens per wave)
__global__ void k_ctx(const __bf16* __restrict__ kb, const __bf16* __restrict__ vb,
                      const __bf16* __restrict__ projS, const unsigned* __restrict__ mg,
                      float* ctx, float* kpsum){
  __shared__ __align__(16) __bf16 kpT[8][16*32];   // per-wave [16 f][32 tok] transpose
  __shared__ float diagS[8][32];
  int wslot = threadIdx.x >> 5, lane = threadIdx.x & 31;
  int wid = (blockIdx.x * blockDim.x + threadIdx.x) >> 5;
  int bh = wid / (Ng/32), ch = wid % (Ng/32);
  int c0 = ch * 32;
  const __bf16* kp_ = kb + ((size_t)bh*Ng + c0)*Dh;
  const __bf16* vp_ = vb + ((size_t)bh*Ng + c0)*Dh;
  float s = 0.f;
  for (int d = 0; d < Dh; ++d){ float kv = (float)kp_[lane*Dh + d]; s += kv*kv; }
  diagS[wslot][lane] = 0.5f * DN2 * s;
  __syncthreads();
  float mval = fdec(*mg);
  v16bf a0  = load_a_bf16(kp_, Dh);
  v16bf a1  = load_a_bf16(kp_ + 16*Dh, Dh);
  v16bf bv0 = load_b_bf16(vp_, Dh);
  v16bf bv1 = load_b_bf16(vp_ + 16, Dh);
  float* ctxp = ctx + (size_t)bh * Fp * Dh;
  for (int ft = 0; ft < 8; ++ft){
    bool valid = (ft*16 + c_col()) < Fq;
    v16bf bP = load_b_sw(projS + ft*512);
    v8f dA = wmma_bf16(a0, bP, zero8());
    v8f dB = wmma_bf16(a1, bP, zero8());
    float colsum = 0.f;
#pragma unroll
    for (int r = 0; r < 8; ++r){
      int tl = c_row(r);
      float e0 = valid ? RATIO*(__expf(dA[r] - diagS[wslot][tl]      - mval) + FEPS) : 0.f;
      float e1 = valid ? RATIO*(__expf(dB[r] - diagS[wslot][16 + tl] - mval) + FEPS) : 0.f;
      kpT[wslot][c_col()*32 + tl]      = (__bf16)e0;
      kpT[wslot][c_col()*32 + 16 + tl] = (__bf16)e1;
      colsum += e0 + e1;
    }
    colsum += __shfl_xor(colsum, 16, 32);
    if (lane < 16) atomicAdd(&kpsum[bh*Fp + ft*16 + lane], colsum);
    __syncthreads();
    v16bf aT = load_a_bf16(&kpT[wslot][0], 32);     // A = kp^T (16 f x 32 tok)
    v8f c0t = wmma_bf16(aT, bv0, zero8());
    v8f c1t = wmma_bf16(aT, bv1, zero8());
#pragma unroll
    for (int r = 0; r < 8; ++r){
      int fr = ft*16 + c_row(r);
      atomicAdd(&ctxp[fr*Dh + c_col()],      c0t[r]);
      atomicAdd(&ctxp[fr*Dh + 16 + c_col()], c1t[r]);
    }
    __syncthreads();
  }
}

// fused qp features + rowmax + d_inv + o = d_inv * (qp @ ctx)
__global__ void k_attn(const __bf16* __restrict__ qb, const __bf16* __restrict__ projS,
                       const __bf16* __restrict__ ctxS, const float* __restrict__ kpsum,
                       const unsigned* __restrict__ mg, __bf16* __restrict__ attn){
  __shared__ __align__(16) __bf16 qpS[8][16*Fp];   // per-wave [16 tok][128 f]
  __shared__ float diagS[8][16];
  int wslot = threadIdx.x >> 5, lane = threadIdx.x & 31;
  int wid = (blockIdx.x * blockDim.x + threadIdx.x) >> 5;
  int bh = wid / (Ng/16), tile = wid % (Ng/16);
  int n0 = tile * 16;
  const __bf16* qp_ = qb + ((size_t)bh*Ng + n0)*Dh;
  if (lane < 16){
    float s = 0.f;
    for (int d = 0; d < Dh; ++d){ float qv = (float)qp_[lane*Dh + d]; s += qv*qv; }
    diagS[wslot][lane] = 0.5f * DN2 * s;
  }
  __syncthreads();
  float mval = fdec(*mg);
  v16bf a = load_a_bf16(qp_, Dh);
  v8f dd[8];
#pragma unroll
  for (int ft = 0; ft < 7; ++ft)
    dd[ft] = wmma_bf16(a, load_b_sw(projS + ft*512), zero8());
  dd[7] = zero8();
  float rm[8];
#pragma unroll
  for (int r = 0; r < 8; ++r) rm[r] = -INFINITY;
#pragma unroll
  for (int ft = 0; ft < 7; ++ft){
    if (ft*16 + c_col() < Fq){
#pragma unroll
      for (int r = 0; r < 8; ++r) rm[r] = fmaxf(rm[r], dd[ft][r]);
    }
  }
#pragma unroll
  for (int off = 1; off < 16; off <<= 1){
#pragma unroll
    for (int r = 0; r < 8; ++r) rm[r] = fmaxf(rm[r], __shfl_xor(rm[r], off, 16));
  }
  float dg[8];
#pragma unroll
  for (int r = 0; r < 8; ++r) dg[r] = diagS[wslot][c_row(r)];
#pragma unroll
  for (int ft = 0; ft < 8; ++ft){
    bool valid = (ft*16 + c_col()) < Fq;
#pragma unroll
    for (int r = 0; r < 8; ++r)
      dd[ft][r] = valid ? RATIO*(__expf(dd[ft][r] - dg[r] - rm[r]) + FEPS) : 0.f;
  }
  float kps[8];
#pragma unroll
  for (int ft = 0; ft < 8; ++ft) kps[ft] = kpsum[bh*Fp + ft*16 + c_col()];
  float dinv[8];
#pragma unroll
  for (int r = 0; r < 8; ++r){
    float p = 0.f;
#pragma unroll
    for (int ft = 0; ft < 8; ++ft) p += dd[ft][r] * kps[ft];
#pragma unroll
    for (int off = 1; off < 16; off <<= 1) p += __shfl_xor(p, off, 16);
    dinv[r] = 1.f / p;
  }
#pragma unroll
  for (int ft = 0; ft < 8; ++ft){
#pragma unroll
    for (int r = 0; r < 8; ++r)
      qpS[wslot][c_row(r)*Fp + ft*16 + c_col()] = (__bf16)dd[ft][r];
  }
  __syncthreads();
  int b_ = bh >> 1, hh = bh & 1;
  const __bf16* ctxp = ctxS + (size_t)bh * 4096;
#pragma unroll
  for (int dt = 0; dt < 2; ++dt){
    v8f oc = zero8();
#pragma unroll
    for (int fc = 0; fc < 4; ++fc){
      v16bf aq = load_a_bf16(&qpS[wslot][fc*32], Fp);
      oc = wmma_bf16(aq, load_b_sw(ctxp + (fc*2 + dt)*512), oc);
    }
#pragma unroll
    for (int r = 0; r < 8; ++r){
      int n = n0 + c_row(r);
      attn[((size_t)b_*Ng + n)*Dm + hh*Dh + dt*16 + c_col()] = (__bf16)(oc[r] * dinv[r]);
    }
  }
}

// h += attn @ wo + bo   (one wave: 16 tokens x all 64 cols)
__global__ void k_wo_res(const __bf16* __restrict__ attn, const __bf16* __restrict__ woS,
                         const float* __restrict__ bo, float* h){
  int wid = (blockIdx.x * blockDim.x + threadIdx.x) >> 5;
  int t0 = wid * 16;
  v8f c[4] = {zero8(), zero8(), zero8(), zero8()};
#pragma unroll
  for (int kt = 0; kt < 2; ++kt){
    v16bf a = load_a_bf16(attn + (size_t)t0*Dm + kt*32, Dm);
#pragma unroll
    for (int nt = 0; nt < 4; ++nt)
      c[nt] = wmma_bf16(a, load_b_sw(woS + (kt*4 + nt)*512), c[nt]);
  }
#pragma unroll
  for (int nt = 0; nt < 4; ++nt){
    int col = nt*16 + c_col();
    float bias = bo[col];
#pragma unroll
    for (int r = 0; r < 8; ++r)
      h[(size_t)(t0 + c_row(r))*Dm + col] += c[nt][r] + bias;
  }
}

// ff = gelu(y @ w1 + b1)   (one wave: 16 tokens x 64-col group; staged b128 store)
__global__ void k_ff1(const __bf16* __restrict__ y, const __bf16* __restrict__ w1S,
                      const float* __restrict__ b1, __bf16* __restrict__ ff){
  __shared__ __align__(16) __bf16 st[8][16*64];
  int wslot = threadIdx.x >> 5, lane = threadIdx.x & 31;
  int wid = (blockIdx.x * blockDim.x + threadIdx.x) >> 5;
  int tile = wid >> 2, ng = wid & 3;
  int t0 = tile * 16, n0g = ng * 64;
  if (t0 + 16 < TOKENS) __builtin_prefetch(y + (size_t)(t0 + 16)*Dm, 0, 1);
  v8f c[4] = {zero8(), zero8(), zero8(), zero8()};
#pragma unroll
  for (int kt = 0; kt < 2; ++kt){
    v16bf a = load_a_bf16(y + (size_t)t0*Dm + kt*32, Dm);
#pragma unroll
    for (int nt = 0; nt < 4; ++nt)
      c[nt] = wmma_bf16(a, load_b_sw(w1S + (kt*16 + ng*4 + nt)*512), c[nt]);
  }
#pragma unroll
  for (int nt = 0; nt < 4; ++nt){
    int lcol = nt*16 + c_col();
    float bias = b1[n0g + lcol];
#pragma unroll
    for (int r = 0; r < 8; ++r){
      float xv = c[nt][r] + bias;
      float g = 0.5f * xv * (1.f + erff(xv * 0.70710678118654752f));  // exact GELU
      st[wslot][c_row(r)*64 + lcol] = (__bf16)g;
    }
  }
  __syncthreads();
#pragma unroll
  for (int i = 0; i < 4; ++i){
    int seg = i*32 + lane;
    int row = seg >> 3, sc = (seg & 7) * 8;
    uint4 d = *(const uint4*)&st[wslot][row*64 + sc];
    *(uint4*)(ff + (size_t)(t0 + row)*Dff + n0g + sc) = d;
  }
}

// h += ff @ w2 + b2   (one wave: 16 tokens x all 64 cols, K=256)
__global__ void k_ff2(const __bf16* __restrict__ ff, const __bf16* __restrict__ w2S,
                      const float* __restrict__ b2, float* h){
  int wid = (blockIdx.x * blockDim.x + threadIdx.x) >> 5;
  int t0 = wid * 16;
  v8f c[4] = {zero8(), zero8(), zero8(), zero8()};
#pragma unroll
  for (int kc = 0; kc < 8; ++kc){
    v16bf a = load_a_bf16(ff + (size_t)t0*Dff + kc*32, Dff);
#pragma unroll
    for (int nt = 0; nt < 4; ++nt)
      c[nt] = wmma_bf16(a, load_b_sw(w2S + (kc*4 + nt)*512), c[nt]);
  }
#pragma unroll
  for (int nt = 0; nt < 4; ++nt){
    int col = nt*16 + c_col();
    float bias = b2[col];
#pragma unroll
    for (int r = 0; r < 8; ++r)
      h[(size_t)(t0 + c_row(r))*Dm + col] += c[nt][r] + bias;
  }
}

__global__ void k_pool(const float* __restrict__ h, float* pooled){
  __shared__ float red[256];
  int b = blockIdx.x >> 6, d = blockIdx.x & 63;
  float s = 0.f;
  for (int n = threadIdx.x; n < Ng; n += 256)
    s += h[((size_t)b*Ng + n)*Dm + d];
  red[threadIdx.x] = s;
  __syncthreads();
  for (int st = 128; st; st >>= 1){
    if ((int)threadIdx.x < st) red[threadIdx.x] += red[threadIdx.x + st];
    __syncthreads();
  }
  if (threadIdx.x == 0) pooled[b*Dm + d] = red[0] * (1.f / Ng);
}

__global__ void k_cls(const float* __restrict__ pooled, const float* __restrict__ cw,
                      const float* __restrict__ cb, float* out){
  int t = threadIdx.x;
  if (t < Bn * 10){
    int b = t / 10, kc = t % 10;
    float s = cb[kc];
    for (int d = 0; d < Dm; ++d) s += pooled[b*Dm + d] * cw[d*10 + kc];
    out[t] = s;
  }
}

// ---------------- host ----------------
extern "C" void kernel_launch(void* const* d_in, const int* in_sizes, int n_in,
                              void* d_out, int out_size, void* d_ws, size_t ws_size,
                              hipStream_t stream){
  (void)in_sizes; (void)n_in; (void)out_size; (void)ws_size;
  const float* x    = (const float*)d_in[0];
  const float* embw = (const float*)d_in[1];
  const float* ln1g = (const float*)d_in[2];
  const float* ln1b = (const float*)d_in[3];
  const float* wq   = (const float*)d_in[4];
  const float* bq   = (const float*)d_in[5];
  const float* wk   = (const float*)d_in[6];
  const float* bk   = (const float*)d_in[7];
  const float* wv   = (const float*)d_in[8];
  const float* bv   = (const float*)d_in[9];
  const float* wo   = (const float*)d_in[10];
  const float* bo   = (const float*)d_in[11];
  const float* proj = (const float*)d_in[12];
  const float* ln2g = (const float*)d_in[13];
  const float* ln2b = (const float*)d_in[14];
  const float* w1   = (const float*)d_in[15];
  const float* b1   = (const float*)d_in[16];
  const float* w2   = (const float*)d_in[17];
  const float* b2   = (const float*)d_in[18];
  const float* clsw = (const float*)d_in[19];
  const float* clsb = (const float*)d_in[20];

  char* wsb = (char*)d_ws;
  size_t off = 0;
  auto take = [&](size_t bytes)->void*{
    void* p = wsb + off; off = (off + bytes + 255) & ~(size_t)255; return p;
  };
  float*    h      = (float*)   take((size_t)TOKENS*Dm*4);
  __bf16*   ybuf   = (__bf16*)  take((size_t)TOKENS*Dm*2);
  __bf16*   qbuf   = (__bf16*)  take((size_t)TOKENS*Dm*2);
  __bf16*   kbuf   = (__bf16*)  take((size_t)TOKENS*Dm*2);
  __bf16*   vbuf   = (__bf16*)  take((size_t)TOKENS*Dm*2);
  __bf16*   attn   = (__bf16*)  take((size_t)TOKENS*Dm*2);
  __bf16*   ffb    = (__bf16*)  take((size_t)TOKENS*Dff*2);
  __bf16*   projS  = (__bf16*)  take((size_t)8*512*2);
  __bf16*   wqS    = (__bf16*)  take((size_t)4*Dm*Dm*2);
  __bf16*   wkS    = (__bf16*)  take((size_t)4*Dm*Dm*2);
  __bf16*   wvS    = (__bf16*)  take((size_t)4*Dm*Dm*2);
  __bf16*   woS    = (__bf16*)  take((size_t)4*Dm*Dm*2);
  __bf16*   w1S    = (__bf16*)  take((size_t)4*Dm*Dff*2);
  __bf16*   w2S    = (__bf16*)  take((size_t)4*Dff*Dm*2);
  float*    ctx    = (float*)   take((size_t)16*Fp*Dh*4);
  __bf16*   ctxS   = (__bf16*)  take((size_t)16*Fp*Dh*2);
  float*    kpsum  = (float*)   take((size_t)16*Fp*4);
  unsigned* mg     = (unsigned*)take(256);
  float*    pooled = (float*)   take((size_t)Bn*Dm*4);

  dim3 blk(256);
  k_embed<<<TOKENS*Dm/256, blk, 0, stream>>>(x, embw, h);
  // pre-swizzle all weights to bf16 B-fragment tiles (all 4 layers at once)
  k_prep_w<<<(4*Dm*Dm)/256,  blk, 0, stream>>>(wq, wqS, Dm,  Dm);
  k_prep_w<<<(4*Dm*Dm)/256,  blk, 0, stream>>>(wk, wkS, Dm,  Dm);
  k_prep_w<<<(4*Dm*Dm)/256,  blk, 0, stream>>>(wv, wvS, Dm,  Dm);
  k_prep_w<<<(4*Dm*Dm)/256,  blk, 0, stream>>>(wo, woS, Dm,  Dm);
  k_prep_w<<<(4*Dm*Dff)/256, blk, 0, stream>>>(w1, w1S, Dm,  Dff);
  k_prep_w<<<(4*Dff*Dm)/256, blk, 0, stream>>>(w2, w2S, Dff, Dm);

  for (int l = 0; l < 4; ++l){
    k_prep_proj<<<(8*512)/256, blk, 0, stream>>>(proj + (size_t)l*Fq*Dh, projS);
    k_ln<<<TOKENS/256, blk, 0, stream>>>(h, ln1g + l*Dm, ln1b + l*Dm, ybuf);
    k_qkv<<<(TOKT*3)/8, blk, 0, stream>>>(ybuf,
        wqS + (size_t)l*Dm*Dm, bq + l*Dm,
        wkS + (size_t)l*Dm*Dm, bk + l*Dm,
        wvS + (size_t)l*Dm*Dm, bv + l*Dm,
        qbuf, kbuf, vbuf);
    k_zero_state<<<(16*Fp*Dh)/256, blk, 0, stream>>>(ctx, kpsum, mg);
    k_dk_max<<<(16*(Ng/16))/8, blk, 0, stream>>>(kbuf, projS, mg);
    k_ctx<<<(16*(Ng/32))/8, blk, 0, stream>>>(kbuf, vbuf, projS, mg, ctx, kpsum);
    k_prep_ctx<<<(16*Fp*Dh)/256, blk, 0, stream>>>(ctx, ctxS);
    k_attn<<<(16*(Ng/16))/8, blk, 0, stream>>>(qbuf, projS, ctxS, kpsum, mg, attn);
    k_wo_res<<<TOKT/8, blk, 0, stream>>>(attn, woS + (size_t)l*Dm*Dm, bo + l*Dm, h);
    k_ln<<<TOKENS/256, blk, 0, stream>>>(h, ln2g + l*Dm, ln2b + l*Dm, ybuf);
    k_ff1<<<(TOKT*4)/8, blk, 0, stream>>>(ybuf, w1S + (size_t)l*Dm*Dff, b1 + l*Dff, ffb);
    k_ff2<<<TOKT/8, blk, 0, stream>>>(ffb, w2S + (size_t)l*Dff*Dm, b2 + l*Dm, h);
  }

  k_pool<<<Bn*Dm, blk, 0, stream>>>(h, pooled);
  k_cls<<<1, 128, 0, stream>>>(pooled, clsw, clsb, (float*)d_out);
}
